// OskarLayer_12987981103315
// MI455X (gfx1250) — matrix operand order repaired and checked by reading, not compiled, not run on hardware
//
#include <hip/hip_runtime.h>

// ---------------------------------------------------------------------------
// Longformer-style layer on gfx1250 (MI455X).
//  - all GEMMs + attention matmuls: v_wmma_f32_16x16x32_bf16 (wave32)
//  - global->LDS staging: global_load_async_to_lds_b128 (ASYNCcnt) with a
//    double-buffered K loop (compute overlaps the next tile's DMA)
//  - GEMM block tile 128x128: 8 waves (4x2), each wave 32x64 = 8 WMMAs/step
// ---------------------------------------------------------------------------

typedef __attribute__((ext_vector_type(16))) __bf16 bf16x16;
typedef __attribute__((ext_vector_type(8)))  __bf16 bf16x8;
typedef __attribute__((ext_vector_type(8)))  float  f32x8;

#define B_   2
#define S_   4096
#define HID_ 768
#define H_   12
#define DH_  64
#define FF_  3072
#define W_   128
#define NT_  (B_ * S_)

// ---------------------------------------------------------------------------
// CDNA5 async global->LDS DMA (inline asm; portable across toolchains)
// ---------------------------------------------------------------------------
__device__ __forceinline__ void async_copy_b128(unsigned lds_off,
                                                const __bf16* g) {
  unsigned long long ga = (unsigned long long)(uintptr_t)g;
  asm volatile("global_load_async_to_lds_b128 %0, %1, off"
               :
               : "v"(lds_off), "v"(ga)
               : "memory");
}
__device__ __forceinline__ void async_wait0() {
  asm volatile("s_wait_asynccnt 0x0" ::: "memory");
}
__device__ __forceinline__ void async_wait4() {
  asm volatile("s_wait_asynccnt 0x4" ::: "memory");
}
__device__ __forceinline__ unsigned lds_addr(const void* p) {
  return (unsigned)(uintptr_t)p;
}

// ---------------------------------------------------------------------------
// WMMA helpers
// ---------------------------------------------------------------------------
__device__ __forceinline__ f32x8 wmma_bf16(bf16x16 a, bf16x16 b, f32x8 c) {
  return __builtin_amdgcn_wmma_f32_16x16x32_bf16(false, a, false, b, (short)0, c,
                                                 false, false);
}

// A-fragment (16x32 bf16). Lane L: M = L&15, K base = (L<16)?0:8,
// elements 0..7 = K base..base+7, elements 8..15 = K base+16..base+23.
__device__ __forceinline__ bf16x16 frag_a(const __bf16* p, int ldk, int lane) {
  const __bf16* q = p + (lane & 15) * ldk + ((lane < 16) ? 0 : 8);
  bf16x8 lo = *reinterpret_cast<const bf16x8*>(q);
  bf16x8 hi = *reinterpret_cast<const bf16x8*>(q + 16);
  return __builtin_shufflevector(lo, hi, 0, 1, 2, 3, 4, 5, 6, 7,
                                 8, 9, 10, 11, 12, 13, 14, 15);
}

// B-fragment (32x16 bf16) from LDS stored [n][k] with stride ldk.
// Lane L: N = L&15, K base = (L<16)?0:8, elements 0..7 = K base..base+7,
// elements 8..15 = K base+16..base+23.
__device__ __forceinline__ bf16x16 frag_b(const __bf16* p, int ldk, int lane) {
  const __bf16* q = p + (lane & 15) * ldk + ((lane < 16) ? 0 : 8);
  bf16x8 lo = *reinterpret_cast<const bf16x8*>(q);
  bf16x8 hi = *reinterpret_cast<const bf16x8*>(q + 16);
  return __builtin_shufflevector(lo, hi, 0, 1, 2, 3, 4, 5, 6, 7,
                                 8, 9, 10, 11, 12, 13, 14, 15);
}

// ---------------------------------------------------------------------------
// f32 -> bf16 conversion (grid-stride)
// ---------------------------------------------------------------------------
__global__ void f32_to_bf16_kernel(const float* __restrict__ in,
                                   __bf16* __restrict__ out, size_t n) {
  size_t i = (size_t)blockIdx.x * blockDim.x + threadIdx.x;
  size_t stride = (size_t)gridDim.x * blockDim.x;
  for (; i < n; i += stride) out[i] = (__bf16)in[i];
}

// f32 (KxN row-major) -> bf16 transposed (NxK row-major).  32x32 LDS tile.
__global__ __launch_bounds__(256)
void f32_to_bf16_t_kernel(const float* __restrict__ in,
                          __bf16* __restrict__ out, int K, int N) {
  __shared__ float tile[32][33];
  const int n0 = blockIdx.x * 32;
  const int k0 = blockIdx.y * 32;
  const int tx = threadIdx.x & 31;
  const int ty = threadIdx.x >> 5;  // 0..7
#pragma unroll
  for (int i = ty; i < 32; i += 8)
    tile[i][tx] = in[(size_t)(k0 + i) * N + n0 + tx];
  __syncthreads();
#pragma unroll
  for (int i = ty; i < 32; i += 8)
    out[(size_t)(n0 + i) * K + k0 + tx] = (__bf16)tile[tx][i];
}

// ---------------------------------------------------------------------------
// bf16 WMMA GEMM:  out = epilogue(A[MxK] @ B[KxN] + bias)
// B supplied pre-transposed (Bt: N x K).  Block tile 128x128, 8 waves (4x2),
// each wave a 32x64 patch (2x4 WMMA tiles, A-fragments reused 4x).
// Double-buffered K loop, tiles filled by async global->LDS DMA.
// ---------------------------------------------------------------------------
__global__ __launch_bounds__(256)
void wmma_gemm_bias_kernel(const __bf16* __restrict__ A,
                           const __bf16* __restrict__ Bt,
                           const float* __restrict__ bias,
                           float* __restrict__ outF,
                           __bf16* __restrict__ outH,
                           int M, int N, int K, float scale, int do_gelu) {
  __shared__ __bf16 sA[2][128 * 32];   // [row][k]
  __shared__ __bf16 sB[2][128 * 32];   // [n][k]

  const int tid  = threadIdx.x;
  const int lane = tid & 31;
  const int wave = tid >> 5;
  const int wm   = wave & 3;    // 0..3  -> 32-row strip
  const int wn   = wave >> 2;   // 0..1  -> 64-col strip
  const int row0 = blockIdx.y * 128;
  const int col0 = blockIdx.x * 128;

  f32x8 acc[2][4] = {};

  auto stage = [&](int kt, int buf) {
    int r  = tid >> 1;
    int ch = (tid & 1) * 16;
    {  // A tile 128x32
      const __bf16* g = A + (size_t)(row0 + r) * K + kt * 32 + ch;
      unsigned l = lds_addr(&sA[buf][r * 32 + ch]);
      async_copy_b128(l, g);
      async_copy_b128(l + 16u, g + 8);
    }
    {  // B tile 128x32 (rows of Bt are contiguous)
      const __bf16* g = Bt + (size_t)(col0 + r) * K + kt * 32 + ch;
      unsigned l = lds_addr(&sB[buf][r * 32 + ch]);
      async_copy_b128(l, g);
      async_copy_b128(l + 16u, g + 8);
    }
  };

  const int nk = K / 32;
  stage(0, 0);
  for (int kt = 0; kt < nk; ++kt) {
    const int cur   = kt & 1;
    const bool more = (kt + 1 < nk);
    if (more) stage(kt + 1, cur ^ 1);     // overlap next fill with compute
    if (more) async_wait4();              // oldest tile (4/thread) complete
    else      async_wait0();
    __syncthreads();

    bf16x16 a0 = frag_a(sA[cur] + (wm * 32 + 0) * 32, 32, lane);
    bf16x16 a1 = frag_a(sA[cur] + (wm * 32 + 16) * 32, 32, lane);
#pragma unroll
    for (int j = 0; j < 4; ++j) {
      bf16x16 b = frag_b(sB[cur] + (wn * 64 + j * 16) * 32, 32, lane);
      acc[0][j] = wmma_bf16(a0, b, acc[0][j]);
      acc[1][j] = wmma_bf16(a1, b, acc[1][j]);
    }
    __syncthreads();
  }

  // ---- epilogue: bias, scale, optional exact GELU; f32 and/or bf16 out ----
#pragma unroll
  for (int i = 0; i < 2; ++i) {
#pragma unroll
    for (int j = 0; j < 4; ++j) {
      int mBase = row0 + wm * 32 + i * 16 + ((lane < 16) ? 0 : 8);
      int n     = col0 + wn * 64 + j * 16 + (lane & 15);
      float bv  = bias ? bias[n] : 0.0f;
#pragma unroll
      for (int r = 0; r < 8; ++r) {
        int m   = mBase + r;
        float v = (acc[i][j][r] + bv) * scale;
        if (do_gelu) v = 0.5f * v * (1.0f + erff(v * 0.70710678118654752f));
        size_t idx = (size_t)m * N + n;
        if (outF) outF[idx] = v;
        if (outH) outH[idx] = (__bf16)v;
      }
    }
  }
}

// ---------------------------------------------------------------------------
// Banded (Longformer) attention, flash-style streaming over the 384-key
// window in 6 slabs of 64.  One block per (chunk, head, batch); 8 waves,
// each wave owns 16 query rows.
// ---------------------------------------------------------------------------
__global__ __launch_bounds__(256)
void band_attn_kernel(const __bf16* __restrict__ qb,
                      const __bf16* __restrict__ kb,
                      const __bf16* __restrict__ vb,
                      const int* __restrict__ am,
                      __bf16* __restrict__ ctxb) {
  const int c    = blockIdx.x;   // chunk 0..31
  const int h    = blockIdx.y;   // head  0..11
  const int bb   = blockIdx.z;   // batch 0..1
  const int tid  = threadIdx.x;
  const int lane = tid & 31;
  const int wave = tid >> 5;

  __shared__ __bf16 sq[128 * 64];     // Q tile, row-major
  __shared__ __bf16 sk[64 * 64];      // K slab, [key][dh]
  __shared__ __bf16 svt[64 * 64];     // V slab transposed, [dh][key]
  __shared__ __bf16 sp[8 * 16 * 64];  // per-wave P staging, [row][key]
  __shared__ float  smsk[64];

  // ---- stage Q tile via async DMA: thread -> (row, 32 dh elems) ----
  {
    int r    = tid >> 1;
    int dp   = (tid & 1) * 32;
    int qpos = c * 128 + r;
    const __bf16* g = qb + (size_t)(bb * S_ + qpos) * HID_ + h * DH_ + dp;
    unsigned l = lds_addr(&sq[r * 64 + dp]);
    async_copy_b128(l, g);
    async_copy_b128(l + 16u, g + 8);
    async_copy_b128(l + 32u, g + 16);
    async_copy_b128(l + 48u, g + 24);
  }
  async_wait0();
  __syncthreads();

  const int m0    = wave * 16;
  const int qpos0 = c * 128 + m0;
  bf16x16 qa0 = frag_a(sq + m0 * 64 + 0, 64, lane);
  bf16x16 qa1 = frag_a(sq + m0 * 64 + 32, 64, lane);

  float m_run[8], l_run[8];
  f32x8 octx[4] = {};
#pragma unroll
  for (int r = 0; r < 8; ++r) { m_run[r] = -1e30f; l_run[r] = 0.0f; }

  for (int j = 0; j < 6; ++j) {
    const int sstart = c * 128 - W_ + j * 64;
    __syncthreads();  // protect sk/svt reuse from previous slab

    // ---- stage K slab [key][dh] (async) + V slab transposed (manual) ----
    {
      int key  = tid >> 2;          // 0..63
      int dp   = (tid & 3) * 16;    // 0,16,32,48
      int kpos = sstart + key;
      bool valid = (kpos >= 0) && (kpos < S_);
      __bf16* dk  = &sk[key * 64 + dp];
      size_t base = (size_t)(bb * S_ + kpos) * HID_ + h * DH_ + dp;
      if (valid) {
        unsigned l = lds_addr(dk);
        async_copy_b128(l, kb + base);        // EXEC-masked per-lane DMA
        async_copy_b128(l + 16u, kb + base + 8);
      } else {
        uint4 z = {0u, 0u, 0u, 0u};
        *reinterpret_cast<uint4*>(dk)     = z;
        *reinterpret_cast<uint4*>(dk + 8) = z;
      }
      __bf16 tv[16];
      if (valid) {
        uint4 v0 = *reinterpret_cast<const uint4*>(vb + base);
        uint4 v1 = *reinterpret_cast<const uint4*>(vb + base + 8);
        *reinterpret_cast<uint4*>(tv)     = v0;
        *reinterpret_cast<uint4*>(tv + 8) = v1;
      } else {
#pragma unroll
        for (int i = 0; i < 16; ++i) tv[i] = (__bf16)0.0f;
      }
#pragma unroll
      for (int i = 0; i < 16; ++i) svt[(dp + i) * 64 + key] = tv[i];
    }
    if (tid < 64) {
      int kpos = sstart + tid;
      smsk[tid] = (kpos < 0 || kpos >= S_)
                      ? -1e30f
                      : (am[bb * S_ + kpos] != 0 ? -10000.0f : 0.0f);
    }
    async_wait0();
    __syncthreads();

    // ---- scores: S = Q @ K^T for 4 column tiles of 16 keys ----
    float sc[4][8];
#pragma unroll
    for (int t = 0; t < 4; ++t) {
      bf16x16 bk0 = frag_b(sk + t * 16 * 64 + 0, 64, lane);
      bf16x16 bk1 = frag_b(sk + t * 16 * 64 + 32, 64, lane);
      f32x8 s = {};
      s = wmma_bf16(qa0, bk0, s);
      s = wmma_bf16(qa1, bk1, s);
      int kpos = sstart + t * 16 + (lane & 15);
      float mk = smsk[t * 16 + (lane & 15)];
#pragma unroll
      for (int r = 0; r < 8; ++r) {
        int qp  = qpos0 + r + ((lane & 16) ? 8 : 0);
        float v = s[r] + mk;
        if (kpos < qp - W_ || kpos > qp + W_) v = -1e30f;
        sc[t][r] = v;
      }
    }

    // ---- online softmax update (half-wave row reductions) ----
#pragma unroll
    for (int r = 0; r < 8; ++r) {
      float mx = fmaxf(fmaxf(sc[0][r], sc[1][r]), fmaxf(sc[2][r], sc[3][r]));
#pragma unroll
      for (int msk = 1; msk < 16; msk <<= 1) mx = fmaxf(mx, __shfl_xor(mx, msk, 32));
      float nm  = fmaxf(m_run[r], mx);
      float scl = __expf(m_run[r] - nm);
      float rs  = 0.0f;
#pragma unroll
      for (int t = 0; t < 4; ++t) {
        float p = __expf(sc[t][r] - nm);
        sc[t][r] = p;
        rs += p;
      }
#pragma unroll
      for (int msk = 1; msk < 16; msk <<= 1) rs += __shfl_xor(rs, msk, 32);
      l_run[r] = l_run[r] * scl + rs;
      m_run[r] = nm;
#pragma unroll
      for (int t = 0; t < 4; ++t) octx[t][r] *= scl;
      int mrow = r + ((lane & 16) ? 8 : 0);
#pragma unroll
      for (int t = 0; t < 4; ++t)
        sp[wave * 1024 + mrow * 64 + t * 16 + (lane & 15)] = (__bf16)sc[t][r];
    }
    __syncthreads();

    // ---- ctx += P @ V ----
    bf16x16 pa0 = frag_a(sp + wave * 1024 + 0, 64, lane);
    bf16x16 pa1 = frag_a(sp + wave * 1024 + 32, 64, lane);
#pragma unroll
    for (int t = 0; t < 4; ++t) {
      bf16x16 bv0 = frag_b(svt + t * 16 * 64 + 0, 64, lane);
      bf16x16 bv1 = frag_b(svt + t * 16 * 64 + 32, 64, lane);
      octx[t] = wmma_bf16(pa0, bv0, octx[t]);
      octx[t] = wmma_bf16(pa1, bv1, octx[t]);
    }
  }

  // ---- normalize and write ctx (layout [b][s][h][dh] = tokens x 768) ----
#pragma unroll
  for (int r = 0; r < 8; ++r) {
    float inv = 1.0f / l_run[r];
    int mrow  = r + ((lane & 16) ? 8 : 0);
    int qpos  = qpos0 + mrow;
#pragma unroll
    for (int t = 0; t < 4; ++t) {
      int d = t * 16 + (lane & 15);
      ctxb[(size_t)(bb * S_ + qpos) * HID_ + h * DH_ + d] =
          (__bf16)(octx[t][r] * inv);
    }
  }
}

// ---------------------------------------------------------------------------
// Fused residual-add + LayerNorm over HID_=768 (one block per row)
// ---------------------------------------------------------------------------
__global__ __launch_bounds__(256)
void add_layernorm_kernel(const float* __restrict__ xa,
                          const float* __restrict__ xb,
                          const float* __restrict__ gamma,
                          const float* __restrict__ beta,
                          float* __restrict__ outF,
                          __bf16* __restrict__ outH) {
  const int row  = blockIdx.x;
  const int tid  = threadIdx.x;
  const int lane = tid & 31;
  const int wave = tid >> 5;
  __shared__ float red[8];

  const float* pa = xa + (size_t)row * HID_;
  const float* pb = xb + (size_t)row * HID_;

  float x[3];
  float s = 0.0f;
#pragma unroll
  for (int i = 0; i < 3; ++i) {
    int idx = tid + i * 256;
    x[i] = pa[idx] + pb[idx];
    s += x[i];
  }
#pragma unroll
  for (int m = 1; m < 32; m <<= 1) s += __shfl_xor(s, m, 32);
  if (lane == 0) red[wave] = s;
  __syncthreads();
  float tot = 0.0f;
#pragma unroll
  for (int i = 0; i < 8; ++i) tot += red[i];
  float mean = tot * (1.0f / (float)HID_);
  __syncthreads();

  float vs = 0.0f;
#pragma unroll
  for (int i = 0; i < 3; ++i) {
    float d = x[i] - mean;
    vs += d * d;
  }
#pragma unroll
  for (int m = 1; m < 32; m <<= 1) vs += __shfl_xor(vs, m, 32);
  if (lane == 0) red[wave] = vs;
  __syncthreads();
  float vtot = 0.0f;
#pragma unroll
  for (int i = 0; i < 8; ++i) vtot += red[i];
  float rstd = rsqrtf(vtot * (1.0f / (float)HID_) + 1e-12f);

#pragma unroll
  for (int i = 0; i < 3; ++i) {
    int idx = tid + i * 256;
    float y = (x[i] - mean) * rstd * gamma[idx] + beta[idx];
    size_t o = (size_t)row * HID_ + idx;
    outF[o] = y;
    if (outH) outH[o] = (__bf16)y;
  }
}

// ---------------------------------------------------------------------------
// Orchestration
// ---------------------------------------------------------------------------
extern "C" void kernel_launch(void* const* d_in, const int* in_sizes, int n_in,
                              void* d_out, int out_size, void* d_ws,
                              size_t ws_size, hipStream_t stream) {
  (void)in_sizes; (void)n_in; (void)out_size; (void)ws_size;

  const float* hs   = (const float*)d_in[0];
  const int*   am   = (const int*)d_in[1];
  const float* Wq   = (const float*)d_in[2];
  const float* bq   = (const float*)d_in[3];
  const float* Wk   = (const float*)d_in[4];
  const float* bk_  = (const float*)d_in[5];
  const float* Wv   = (const float*)d_in[6];
  const float* bv_  = (const float*)d_in[7];
  const float* Wd   = (const float*)d_in[8];
  const float* bd   = (const float*)d_in[9];
  const float* ln1g = (const float*)d_in[10];
  const float* ln1b = (const float*)d_in[11];
  const float* Wf   = (const float*)d_in[12];
  const float* bf_  = (const float*)d_in[13];
  const float* Wo   = (const float*)d_in[14];
  const float* bo   = (const float*)d_in[15];
  const float* ln2g = (const float*)d_in[16];
  const float* ln2b = (const float*)d_in[17];

  char* wsp = (char*)d_ws;
  size_t off = 0;
  auto alloc = [&](size_t bytes) -> void* {
    off = (off + 255) & ~(size_t)255;
    void* p = wsp + off;
    off += bytes;
    return p;
  };

  const size_t TOK = (size_t)NT_;
  __bf16* hsb   = (__bf16*)alloc(TOK * HID_ * 2);
  __bf16* wqT   = (__bf16*)alloc((size_t)HID_ * HID_ * 2);  // N x K
  __bf16* wkT   = (__bf16*)alloc((size_t)HID_ * HID_ * 2);
  __bf16* wvT   = (__bf16*)alloc((size_t)HID_ * HID_ * 2);
  __bf16* wdT   = (__bf16*)alloc((size_t)HID_ * HID_ * 2);
  __bf16* wfT   = (__bf16*)alloc((size_t)HID_ * FF_ * 2);   // 3072 x 768
  __bf16* woT   = (__bf16*)alloc((size_t)FF_ * HID_ * 2);   // 768 x 3072
  __bf16* qbuf  = (__bf16*)alloc(TOK * HID_ * 2);
  __bf16* kbuf  = (__bf16*)alloc(TOK * HID_ * 2);
  __bf16* vbuf  = (__bf16*)alloc(TOK * HID_ * 2);
  __bf16* ctxb  = (__bf16*)alloc(TOK * HID_ * 2);
  float*  projF = (float*)alloc(TOK * HID_ * 4);
  float*  attnF = (float*)alloc(TOK * HID_ * 4);
  __bf16* attnB = (__bf16*)alloc(TOK * HID_ * 2);
  __bf16* ff1B  = (__bf16*)alloc(TOK * FF_ * 2);
  float*  ffF   = (float*)alloc(TOK * HID_ * 4);

  // 1) convert activations; convert + transpose weights (K x N -> N x K)
  f32_to_bf16_kernel<<<2048, 256, 0, stream>>>(hs, hsb, TOK * HID_);
  dim3 gT(HID_ / 32, HID_ / 32);
  f32_to_bf16_t_kernel<<<gT, 256, 0, stream>>>(Wq, wqT, HID_, HID_);
  f32_to_bf16_t_kernel<<<gT, 256, 0, stream>>>(Wk, wkT, HID_, HID_);
  f32_to_bf16_t_kernel<<<gT, 256, 0, stream>>>(Wv, wvT, HID_, HID_);
  f32_to_bf16_t_kernel<<<gT, 256, 0, stream>>>(Wd, wdT, HID_, HID_);
  f32_to_bf16_t_kernel<<<dim3(FF_ / 32, HID_ / 32), 256, 0, stream>>>(
      Wf, wfT, HID_, FF_);
  f32_to_bf16_t_kernel<<<dim3(HID_ / 32, FF_ / 32), 256, 0, stream>>>(
      Wo, woT, FF_, HID_);

  // 2) QKV projections (Q pre-scaled by 1/sqrt(DH))
  dim3 gQ(HID_ / 128, NT_ / 128);
  wmma_gemm_bias_kernel<<<gQ, 256, 0, stream>>>(hsb, wqT, bq, nullptr, qbuf,
                                                NT_, HID_, HID_, 0.125f, 0);
  wmma_gemm_bias_kernel<<<gQ, 256, 0, stream>>>(hsb, wkT, bk_, nullptr, kbuf,
                                                NT_, HID_, HID_, 1.0f, 0);
  wmma_gemm_bias_kernel<<<gQ, 256, 0, stream>>>(hsb, wvT, bv_, nullptr, vbuf,
                                                NT_, HID_, HID_, 1.0f, 0);

  // 3) banded attention
  band_attn_kernel<<<dim3(S_ / 128, H_, B_), 256, 0, stream>>>(qbuf, kbuf, vbuf,
                                                               am, ctxb);

  // 4) output projection
  wmma_gemm_bias_kernel<<<gQ, 256, 0, stream>>>(ctxb, wdT, bd, projF, nullptr,
                                                NT_, HID_, HID_, 1.0f, 0);

  // 5) residual + LN1
  add_layernorm_kernel<<<NT_, 256, 0, stream>>>(hs, projF, ln1g, ln1b, attnF,
                                                attnB);

  // 6) FFN: GEMM+GELU, then GEMM
  dim3 gF1(FF_ / 128, NT_ / 128);
  wmma_gemm_bias_kernel<<<gF1, 256, 0, stream>>>(attnB, wfT, bf_, nullptr, ff1B,
                                                 NT_, FF_, HID_, 1.0f, 1);
  wmma_gemm_bias_kernel<<<gQ, 256, 0, stream>>>(ff1B, woT, bo, ffF, nullptr,
                                                NT_, HID_, FF_, 1.0f, 0);

  // 7) residual + LN2 -> final f32 output
  add_layernorm_kernel<<<NT_, 256, 0, stream>>>(ffF, attnF, ln2g, ln2b,
                                                (float*)d_out, nullptr);
}
